// eGNN_60911226192247
// MI455X (gfx1250) — compile-verified, matrix operand order checked by reading, not run on hardware
//
#include <hip/hip_runtime.h>
#include <hip/hip_bf16.h>
#include <math.h>

// ---------------------------------------------------------------------------
// eGNN on gfx1250: per-edge MLPs fused into f16 WMMA GEMMs (f32 accumulate),
// fragment-major (pre-swizzled) B weights -> 2x global_load_b128 per fragment,
// scatter-mean via native f32 global atomics, wave32 16-edge tiles.
// ---------------------------------------------------------------------------

typedef __attribute__((ext_vector_type(16))) _Float16 v16h;
typedef __attribute__((ext_vector_type(8)))  _Float16 v8h;
typedef __attribute__((ext_vector_type(8)))  float    v8f;

#define NATOM 50000
#define NEDGE 800000
#define NGRAPH 64
#define INV_SQRT_MMSG 0.17677669529663687f

__device__ __forceinline__ float lrelu(float x) { return x > 0.f ? x : 0.01f * x; }
__device__ __forceinline__ float sigm(float x)  { return 1.f / (1.f + __expf(-x)); }

__device__ __forceinline__ v8f wmma16(v16h a, v16h b, v8f c) {
  // D = A(16x32 f16) * B(32x16 f16) + C(16x16 f32)
  return __builtin_amdgcn_wmma_f32_16x16x32_f16(false, a, false, b, (short)0, c,
                                                false, false);
}

// A-fragment from an LDS row: lane (ln&15) holds row M; base=(ln>=16)?8:0.
// Two aligned 16B chunks concatenated -> minimal repacking.
__device__ __forceinline__ v16h loadA(const _Float16* row, int k0, int base) {
  v8h lo = *(const v8h*)(row + k0 + base);
  v8h hi = *(const v8h*)(row + k0 + base + 16);
  return __builtin_shufflevector(lo, hi, 0, 1, 2, 3, 4, 5, 6, 7,
                                 8, 9, 10, 11, 12, 13, 14, 15);
}
// B-fragment from fragment-major packed weights: per (frag,lane) 16 halves
// contiguous -> two global_load_b128, coalesced across the wave.
__device__ __forceinline__ v16h loadBfrag(const _Float16* __restrict__ P,
                                          int frag, int ln) {
  const _Float16* p = P + ((frag << 5) + ln) * 16;
  v8h lo = *(const v8h*)p;
  v8h hi = *(const v8h*)(p + 8);
  return __builtin_shufflevector(lo, hi, 0, 1, 2, 3, 4, 5, 6, 7,
                                 8, 9, 10, 11, 12, 13, 14, 15);
}

__device__ __forceinline__ void atomicMaxF(float* p, float v) {
  if (v >= 0.f) atomicMax((int*)p, __float_as_int(v));
  else          atomicMin((unsigned int*)p, (unsigned int)__float_as_int(v));
}
__device__ __forceinline__ void atomAdd(float* p, float v) { unsafeAtomicAdd(p, v); }

// ------------------------------- utility kernels ---------------------------
__global__ void fill_f32(float* p, int n, float v) {
  int i = blockIdx.x * blockDim.x + threadIdx.x;
  if (i < n) p[i] = v;
}
__global__ void fill_f16(_Float16* p, int n) {
  int i = blockIdx.x * blockDim.x + threadIdx.x;
  if (i < n) p[i] = (_Float16)0.f;
}
__global__ void pack_f16(_Float16* dst, int dstStride, int rowOff, int colOff,
                         const float* src, int K, int Ncols) {
  int idx = blockIdx.x * blockDim.x + threadIdx.x;
  if (idx >= K * Ncols) return;
  int k = idx / Ncols, n = idx % Ncols;
  dst[(rowOff + k) * dstStride + colOff + n] = (_Float16)src[idx];
}
// Row-major [K][N] f16 -> fragment-major: frag=(nt*ksteps+ks), then 32 lanes
// x 16 halves matching the v_wmma_f32_16x16x32_f16 B operand layout.
__global__ void bswizzle(_Float16* __restrict__ dst,
                         const _Float16* __restrict__ src, int K, int N) {
  int idx = blockIdx.x * blockDim.x + threadIdx.x;
  int total = (K / 32) * (N / 16) * 512;
  if (idx >= total) return;
  int t    = idx & 15;
  int ln   = (idx >> 4) & 31;
  int frag = idx >> 9;
  int ksteps = K / 32;
  int nt = frag / ksteps, ks = frag % ksteps;
  int base = (ln >= 16) ? 8 : 0;
  int n = nt * 16 + (ln & 15);
  int kk = ks * 32 + base + ((t < 8) ? t : (8 + t));  // t>=8 -> base+16+(t-8)
  dst[idx] = src[kk * N + n];
}

// -------- fuse input linears into stage-1 weight: u[64] -> [sig|msg|q][96] ---
__global__ void prep_w1(_Float16* W1f,
                        const float* LnEw, const float* LnKw,
                        const float* LnQw, const float* LnVw,
                        const float* sg0w, const float* pm0w) {
  int idx = blockIdx.x * blockDim.x + threadIdx.x;
  if (idx >= 64 * 96) return;
  int k = idx / 96, n = idx % 96;
  float v = 0.f;
  if (n < 32) {                                   // sigma hidden
    if (k < 21)            { for (int t = 0; t < 21; ++t) v += LnEw[k*21+t] * sg0w[t*32+n]; }
    else if (k >= 39 && k < 57) { int kk = k-39; for (int t = 0; t < 18; ++t) v += LnKw[kk*18+t] * sg0w[(21+t)*32+n]; }
  } else if (n < 64) {                            // phi_msg hidden
    int nn = n - 32;
    if (k < 21)      { for (int t = 0; t < 21; ++t) v += LnEw[k*21+t] * pm0w[t*32+nn]; }
    else if (k < 39) { int kk = k-21; for (int t = 0; t < 18; ++t) v += LnVw[kk*18+t] * pm0w[(21+t)*32+nn]; }
    else if (k < 57) { int kk = k-39; for (int t = 0; t < 18; ++t) v += LnVw[kk*18+t] * pm0w[(39+t)*32+nn]; }
  } else {                                        // q = LnQ(xi)
    int nn = n - 64;
    if (k >= 21 && k < 39) v = LnQw[(k-21)*32+nn];
  }
  W1f[idx] = (_Float16)v;
}

__global__ void prep_b1(float* b1f,
                        const float* LnEb, const float* LnKb,
                        const float* LnQb, const float* LnVb,
                        const float* sg0w, const float* sg0b,
                        const float* pm0w, const float* pm0b) {
  int n = threadIdx.x;
  if (n >= 96) return;
  float v;
  if (n < 32) {
    v = sg0b[n];
    for (int t = 0; t < 21; ++t) v += LnEb[t] * sg0w[t*32+n];
    for (int t = 0; t < 18; ++t) v += LnKb[t] * sg0w[(21+t)*32+n];
  } else if (n < 64) {
    int nn = n - 32; v = pm0b[nn];
    for (int t = 0; t < 21; ++t) v += LnEb[t] * pm0w[t*32+nn];
    for (int t = 0; t < 18; ++t) v += LnVb[t] * (pm0w[(21+t)*32+nn] + pm0w[(39+t)*32+nn]);
  } else v = LnQb[n - 64];
  b1f[n] = v;
}

// ------------------------------- edge kernel -------------------------------
// 128 threads = 4 waves; each wave owns 16 edges; 28 WMMAs per wave-tile.
__global__ void __launch_bounds__(128)
edge_kernel(const float* __restrict__ x, const float* __restrict__ chi,
            const int* __restrict__ ei, const float* __restrict__ ea,
            const float* __restrict__ ev,
            const _Float16* __restrict__ W1p, const float* __restrict__ b1f,
            const _Float16* __restrict__ W2p, const float* __restrict__ sg1b,
            const float* __restrict__ pm1b,
            const _Float16* __restrict__ W3p, const float* __restrict__ al0b,
            const float* __restrict__ be0b,
            const float* __restrict__ al1w, const float* __restrict__ al1b,
            const float* __restrict__ be1w, const float* __restrict__ be1b,
            float* __restrict__ agg, float* __restrict__ cnt) {
  __shared__ _Float16 sU[4][16][64];   // [ea|xi|xj|pad] rows
  __shared__ _Float16 sT[4][16][64];   // tempv rows
  __shared__ _Float16 sH[4][16][64];   // stage-1 hidden (lrelu, f16)
  __shared__ float    sO[4][16][96];   // stage-2 out (0..63) + q (64..95)

  const int tid = threadIdx.x;
  const int wv = tid >> 5, ln = tid & 31;
  const int er = ln & 15, hh = ln >> 4;
  const int e = blockIdx.x * 64 + wv * 16 + er;
  const bool valid = e < NEDGE;

  int jj = 0, ii = 0;
  float ev0=0,ev1=0,ev2=0, ci0=0,ci1=0,ci2=0, cj0=0,cj1=0,cj2=0;
  if (valid) {
    jj = ei[e]; ii = ei[NEDGE + e];
    ev0 = ev[e*3+0]; ev1 = ev[e*3+1]; ev2 = ev[e*3+2];
    ci0 = chi[ii*3+0]; ci1 = chi[ii*3+1]; ci2 = chi[ii*3+2];
    cj0 = chi[jj*3+0]; cj1 = chi[jj*3+1]; cj2 = chi[jj*3+2];
  }
  for (int c = hh*32; c < hh*32 + 32; ++c) {      // stage rows (2 lanes/edge)
    float v = 0.f;
    if (valid) {
      if (c < 21)      v = ea[e*21 + c];
      else if (c < 39) v = x[ii*18 + (c-21)];
      else if (c < 57) v = x[jj*18 + (c-39)];
    }
    sU[wv][er][c] = (_Float16)v;
  }
  for (int c = hh*32; c < hh*32 + 32; ++c) {      // tempv: [6 scalars|xj|xi|pad]
    float v = 0.f;
    if (valid) {
      if (c == 0) v = ev0*ci0 + ev1*ci1 + ev2*ci2;
      else if (c == 1) v = ev0*cj0 + ev1*cj1 + ev2*cj2;
      else if (c == 2) v = ci0*cj0 + ci1*cj1 + ci2*cj2;
      else if (c == 3) v = sqrtf(ev0*ev0 + ev1*ev1 + ev2*ev2);
      else if (c == 4) v = sqrtf(ci0*ci0 + ci1*ci1 + ci2*ci2);
      else if (c == 5) v = sqrtf(cj0*cj0 + cj1*cj1 + cj2*cj2);
      else if (c < 24) v = x[jj*18 + (c-6)];
      else if (c < 42) v = x[ii*18 + (c-24)];
    }
    sT[wv][er][c] = (_Float16)v;
  }
  __syncthreads();

  const int base = hh * 8;
  // ---- GEMM1: u[16x64] x W1[64x96] -> sigma_h | msg_h | q  (frag = nt*2+ks)
  v16h a0 = loadA(&sU[wv][er][0], 0, base);
  v16h a1 = loadA(&sU[wv][er][0], 32, base);
#pragma unroll
  for (int nt = 0; nt < 6; ++nt) {
    const int n = nt * 16 + er;
    v8f c = {};
    c = wmma16(a0, loadBfrag(W1p, nt * 2 + 0, ln), c);
    c = wmma16(a1, loadBfrag(W1p, nt * 2 + 1, ln), c);
    const float bias = b1f[n];
#pragma unroll
    for (int r = 0; r < 8; ++r) {
      const float v = c[r] + bias;
      if (n < 64) sH[wv][base + r][n] = (_Float16)lrelu(v);
      else        sO[wv][base + r][n] = v;          // q (linear)
    }
  }
  __syncthreads();
  // ---- GEMM2: lrelu(hidden)[16x64] x [sigma2|m2][64x64]
  v16h h0 = loadA(&sH[wv][er][0], 0, base);
  v16h h1 = loadA(&sH[wv][er][0], 32, base);
#pragma unroll
  for (int nt = 0; nt < 4; ++nt) {
    const int n = nt * 16 + er;
    v8f c = {};
    c = wmma16(h0, loadBfrag(W2p, nt * 2 + 0, ln), c);
    c = wmma16(h1, loadBfrag(W2p, nt * 2 + 1, ln), c);
    const float bias = (n < 32) ? sg1b[n] : pm1b[n - 32];
#pragma unroll
    for (int r = 0; r < 8; ++r) sO[wv][base + r][n] = c[r] + bias;
  }
  __syncthreads();
  // ---- epilogue A: w = sigmoid(q * lrelu(sig) / sqrt(32)); scatter w*msg
  if (valid) {
#pragma unroll
    for (int c0 = 0; c0 < 16; ++c0) {
      const int c  = hh * 16 + c0;
      const float q  = sO[wv][er][64 + c];
      const float sg = lrelu(sO[wv][er][c]);
      const float mg = lrelu(sO[wv][er][32 + c]);
      atomAdd(&agg[ii*36 + c], sigm(q * sg * INV_SQRT_MMSG) * mg);
    }
  }
  __syncthreads();
  // ---- GEMM3: tempv[16x64] x [alpha0|beta0][64x64] -> lrelu'd hidden
  v16h t0 = loadA(&sT[wv][er][0], 0, base);
  v16h t1 = loadA(&sT[wv][er][0], 32, base);
#pragma unroll
  for (int nt = 0; nt < 4; ++nt) {
    const int n = nt * 16 + er;
    v8f c = {};
    c = wmma16(t0, loadBfrag(W3p, nt * 2 + 0, ln), c);
    c = wmma16(t1, loadBfrag(W3p, nt * 2 + 1, ln), c);
    const float bias = (n < 32) ? al0b[n] : be0b[n - 32];
#pragma unroll
    for (int r = 0; r < 8; ++r) sO[wv][base + r][n] = lrelu(c[r] + bias);
  }
  __syncthreads();
  // ---- epilogue B: a*edge_vec + sigmoid(b)*chi_j, scatter + count
  if (hh == 0 && valid) {
    float aacc = al1b[0], bacc = be1b[0];
    for (int c2 = 0; c2 < 32; ++c2) {
      aacc += al1w[c2] * sO[wv][er][c2];
      bacc += be1w[c2] * sO[wv][er][32 + c2];
    }
    const float bg = sigm(bacc);
    atomAdd(&agg[ii*36 + 32], aacc * ev0 + bg * cj0);
    atomAdd(&agg[ii*36 + 33], aacc * ev1 + bg * cj1);
    atomAdd(&agg[ii*36 + 34], aacc * ev2 + bg * cj2);
    atomAdd(&cnt[ii], 1.0f);
  }
}

// ------------------------- node phi_out (WMMA) -----------------------------
__global__ void __launch_bounds__(128)
node_phi_out(const float* __restrict__ agg, const float* __restrict__ cnt,
             const float* __restrict__ x,
             const _Float16* __restrict__ Wo1p, const float* __restrict__ po0b,
             const _Float16* __restrict__ Wo2p, const float* __restrict__ po1b,
             float* __restrict__ xout, float* __restrict__ chiout) {
  __shared__ _Float16 sU[4][16][64];
  __shared__ _Float16 sH[4][16][32];
  const int tid = threadIdx.x, wv = tid >> 5, ln = tid & 31;
  const int er = ln & 15, hh = ln >> 4;
  const int nd = blockIdx.x * 64 + wv * 16 + er;
  const bool valid = nd < NATOM;
  float inv = 1.f;
  if (valid) inv = 1.f / fmaxf(cnt[nd], 1.f);
  for (int c = hh*32; c < hh*32 + 32; ++c) {
    float v = 0.f;
    if (valid) {
      if (c < 32)      v = agg[nd*36 + c] * inv;
      else if (c < 50) v = x[nd*18 + (c-32)];
    }
    sU[wv][er][c] = (_Float16)v;
  }
  if (valid && hh == 0) {
#pragma unroll
    for (int d = 0; d < 3; ++d) chiout[nd*3 + d] = agg[nd*36 + 32 + d] * inv;
  }
  __syncthreads();
  const int base = hh * 8;
  v16h a0 = loadA(&sU[wv][er][0], 0, base);
  v16h a1 = loadA(&sU[wv][er][0], 32, base);
#pragma unroll
  for (int nt = 0; nt < 2; ++nt) {
    const int n = nt * 16 + er;
    v8f c = {};
    c = wmma16(a0, loadBfrag(Wo1p, nt * 2 + 0, ln), c);
    c = wmma16(a1, loadBfrag(Wo1p, nt * 2 + 1, ln), c);
#pragma unroll
    for (int r = 0; r < 8; ++r)
      sH[wv][base + r][n] = (_Float16)lrelu(c[r] + po0b[n]);
  }
  __syncthreads();
  v16h h0 = loadA(&sH[wv][er][0], 0, base);
#pragma unroll
  for (int nt = 0; nt < 2; ++nt) {
    const int n = nt * 16 + er;
    v8f c = {};
    c = wmma16(h0, loadBfrag(Wo2p, nt, ln), c);   // K=32 -> frag = nt
    if (n < 18) {
#pragma unroll
      for (int r = 0; r < 8; ++r) {
        const int node = blockIdx.x * 64 + wv * 16 + base + r;
        if (node < NATOM) xout[node*18 + n] = c[r] + po1b[n];
      }
    }
  }
}

// ---------------------- node mlp3 (phi/psi) + pooled reduce ----------------
__global__ void __launch_bounds__(128)
node_mlp3(const float* __restrict__ x,
          const _Float16* __restrict__ Wm1p, const float* __restrict__ b1,
          const _Float16* __restrict__ Wm2p, const float* __restrict__ b2,
          const float* __restrict__ w3, const float* __restrict__ b3,
          const int* __restrict__ batch, float* __restrict__ outp, int mode) {
  __shared__ _Float16 sU[4][16][32];
  __shared__ _Float16 sH[4][16][64];
  __shared__ float    sO2[4][16][64];
  const int tid = threadIdx.x, wv = tid >> 5, ln = tid & 31;
  const int er = ln & 15, hh = ln >> 4;
  const int nd = blockIdx.x * 64 + wv * 16 + er;
  const bool valid = nd < NATOM;
  for (int c = hh*16; c < hh*16 + 16; ++c) {
    float v = (valid && c < 18) ? x[nd*18 + c] : 0.f;
    sU[wv][er][c] = (_Float16)v;
  }
  __syncthreads();
  const int base = hh * 8;
  v16h a0 = loadA(&sU[wv][er][0], 0, base);       // K = 32 (x padded)
#pragma unroll
  for (int nt = 0; nt < 4; ++nt) {
    const int n = nt * 16 + er;
    v8f c = {};
    c = wmma16(a0, loadBfrag(Wm1p, nt, ln), c);   // K=32 -> frag = nt
#pragma unroll
    for (int r = 0; r < 8; ++r)
      sH[wv][base + r][n] = (_Float16)lrelu(c[r] + b1[n]);
  }
  __syncthreads();
  v16h h0 = loadA(&sH[wv][er][0], 0, base);
  v16h h1 = loadA(&sH[wv][er][0], 32, base);
#pragma unroll
  for (int nt = 0; nt < 4; ++nt) {
    const int n = nt * 16 + er;
    v8f c = {};
    c = wmma16(h0, loadBfrag(Wm2p, nt * 2 + 0, ln), c);
    c = wmma16(h1, loadBfrag(Wm2p, nt * 2 + 1, ln), c);
#pragma unroll
    for (int r = 0; r < 8; ++r)
      sO2[wv][base + r][n] = lrelu(c[r] + b2[n]);
  }
  __syncthreads();
  if (hh == 0 && valid) {
    float y = b3[0];
    for (int c = 0; c < 64; ++c) y += w3[c] * sO2[wv][er][c];
    const int g = batch[nd];
    if (mode == 0) atomicMaxF(&outp[g], y);
    else           atomAdd(&outp[g], y);
  }
}

// ----------------------------- pooling helpers -----------------------------
__global__ void graph_count(const int* batch, float* gcnt) {
  int v = blockIdx.x * blockDim.x + threadIdx.x;
  if (v < NATOM) atomAdd(&gcnt[batch[v]], 1.f);
}
__global__ void pool_ft_ch(const float* x, const float* chi, const int* batch,
                           float* ft, float* ch, int slot) {
  int v = blockIdx.x * blockDim.x + threadIdx.x;
  if (v >= NATOM) return;
  int g = batch[v];
  for (int c = 0; c < 18; ++c) atomAdd(&ft[g*54 + slot*18 + c], x[v*18 + c]);
  for (int d = 0; d < 3;  ++d) atomAdd(&ch[g*9  + slot*3  + d], chi[v*3 + d]);
}

// ------------------------------ readout head -------------------------------
__global__ void final_kernel(const float* ft, const float* ch, const float* gcnt,
                             const float* atmax, const float* atmean,
                             const float* m0w, const float* m0b,
                             const float* m1w, const float* m1b,
                             const float* m2w, const float* m2b,
                             const float* m3w, const float* m3b,
                             const float* r0w, const float* r0b,
                             const float* r1w, const float* r1b,
                             float* out) {
  __shared__ float sA[64][64];
  __shared__ float sB[64][128];
  const int g = threadIdx.x;
  if (g >= NGRAPH) return;
  const float inv = 1.f / fmaxf(gcnt[g], 1.f);
  float chv[9];
  for (int t = 0; t < 9; ++t) chv[t] = ch[g*9 + t] * inv;
  for (int l = 0; l < 3; ++l)
    for (int k = 0; k < 3; ++k) {
      float s = 0.f;
      for (int d = 0; d < 3; ++d) s += chv[l*3 + d] * chv[k*3 + d];
      sA[g][l*3 + k] = s;                            // gram (9)
    }
  for (int t = 0; t < 54; ++t) sA[g][9 + t] = ft[g*54 + t] * inv;  // ftc (54)
  for (int n = 0; n < 128; ++n) {
    float s = m0b[n];
    for (int k = 0; k < 63; ++k) s += sA[g][k] * m0w[k*128 + n];
    sB[g][n] = lrelu(s);
  }
  for (int n = 0; n < 64; ++n) {
    float s = m1b[n];
    for (int k = 0; k < 128; ++k) s += sB[g][k] * m1w[k*64 + n];
    sA[g][n] = lrelu(s);
  }
  for (int n = 0; n < 32; ++n) {
    float s = m2b[n];
    for (int k = 0; k < 64; ++k) s += sA[g][k] * m2w[k*32 + n];
    sB[g][n] = lrelu(s);
  }
  float atsum = m3b[0];
  for (int k = 0; k < 32; ++k) atsum += sB[g][k] * m3w[k];
  float red[7];
  red[0] = atmax[g];          red[1] = atmax[64 + g];        red[2] = atmax[128 + g];
  red[3] = atmean[g] * inv;   red[4] = atmean[64 + g] * inv; red[5] = atmean[128 + g] * inv;
  red[6] = atsum;
  float acc = r1b[0];
  for (int n = 0; n < 16; ++n) {
    float s = r0b[n];
    for (int k = 0; k < 7; ++k) s += red[k] * r0w[k*16 + n];
    acc += (s * sigm(s)) * r1w[n];
  }
  out[g] = acc;
}

// ------------------------------- launcher ----------------------------------
// Param leaf indexing assumes: top-level insertion order [x, edge_attr,
// edge_vec, <params leaves>, edge_index, batch] with params flattened in JAX
// order (dict keys sorted, lists in order, each _mk dict -> b then w).
extern "C" void kernel_launch(void* const* d_in, const int* in_sizes, int n_in,
                              void* d_out, int out_size, void* d_ws, size_t ws_size,
                              hipStream_t stream) {
  (void)in_sizes; (void)n_in; (void)out_size; (void)ws_size;
  auto F = [&](int i) { return (const float*)d_in[i]; };
  const float* x0 = F(0);
  const float* ea = F(1);
  const float* ev = F(2);
  const int P0 = 3;
  const int* ei    = (const int*)d_in[P0 + 132];
  const int* batch = (const int*)d_in[P0 + 133];

  // ---- workspace bump allocator
  char* wp = (char*)d_ws;
  auto alloc = [&](size_t b) { void* p = (void*)wp; wp += (b + 255) & ~(size_t)255; return p; };
  float* x1 = (float*)alloc((size_t)NATOM * 18 * 4);
  float* x2 = (float*)alloc((size_t)NATOM * 18 * 4);
  float* x3 = (float*)alloc((size_t)NATOM * 18 * 4);
  float* chi0 = (float*)alloc((size_t)NATOM * 3 * 4);
  float* chi1 = (float*)alloc((size_t)NATOM * 3 * 4);
  float* chi2 = (float*)alloc((size_t)NATOM * 3 * 4);
  float* chi3 = (float*)alloc((size_t)NATOM * 3 * 4);
  float* agg = (float*)alloc((size_t)NATOM * 36 * 4);
  float* cnt = (float*)alloc((size_t)NATOM * 4);
  _Float16 *W1p[3], *W2p[3], *W3p[3], *Wo1p[3], *Wo2p[3];
  float* b1f[3];
  for (int l = 0; l < 3; ++l) {
    W1p[l]  = (_Float16*)alloc(64 * 96 * 2);
    b1f[l]  = (float*)alloc(96 * 4);
    W2p[l]  = (_Float16*)alloc(64 * 64 * 2);
    W3p[l]  = (_Float16*)alloc(64 * 64 * 2);
    Wo1p[l] = (_Float16*)alloc(64 * 32 * 2);
    Wo2p[l] = (_Float16*)alloc(32 * 32 * 2);
  }
  _Float16 *Wm1p[6], *Wm2p[6];                    // 0..2 phi, 3..5 psi
  for (int s = 0; s < 6; ++s) {
    Wm1p[s] = (_Float16*)alloc(32 * 64 * 2);
    Wm2p[s] = (_Float16*)alloc(64 * 64 * 2);
  }
  _Float16* tmp = (_Float16*)alloc(64 * 96 * 2);  // row-major staging
  float* ft     = (float*)alloc(64 * 54 * 4);
  float* chp    = (float*)alloc(64 * 9 * 4);
  float* atmax  = (float*)alloc(192 * 4);
  float* atmean = (float*)alloc(192 * 4);
  float* gcnt   = (float*)alloc(64 * 4);

  // ---- per-layer weight fusion (f32 -> row-major f16 -> fragment-major)
  for (int l = 0; l < 3; ++l) {
    const int B = P0 + 28 * l;
    const float *LnEb=F(B+0), *LnEw=F(B+1), *LnKb=F(B+2), *LnKw=F(B+3),
                *LnQb=F(B+4), *LnQw=F(B+5), *LnVb=F(B+6), *LnVw=F(B+7),
                *al0w=F(B+9), *be0w=F(B+13),
                *pm0b=F(B+16), *pm0w=F(B+17), *pm1w=F(B+19),
                *po0w=F(B+21), *po1w=F(B+23),
                *sg0b=F(B+24), *sg0w=F(B+25), *sg1w=F(B+27);
    prep_w1<<<24, 256, 0, stream>>>(tmp, LnEw, LnKw, LnQw, LnVw, sg0w, pm0w);
    bswizzle<<<24, 256, 0, stream>>>(W1p[l], tmp, 64, 96);
    prep_b1<<<1, 96, 0, stream>>>(b1f[l], LnEb, LnKb, LnQb, LnVb, sg0w, sg0b, pm0w, pm0b);
    fill_f16<<<16, 256, 0, stream>>>(tmp, 4096);
    pack_f16<<<4, 256, 0, stream>>>(tmp, 64, 0, 0, sg1w, 32, 32);
    pack_f16<<<4, 256, 0, stream>>>(tmp, 64, 32, 32, pm1w, 32, 32);
    bswizzle<<<16, 256, 0, stream>>>(W2p[l], tmp, 64, 64);
    fill_f16<<<16, 256, 0, stream>>>(tmp, 4096);
    pack_f16<<<6, 256, 0, stream>>>(tmp, 64, 0, 0, al0w, 42, 32);
    pack_f16<<<6, 256, 0, stream>>>(tmp, 64, 0, 32, be0w, 42, 32);
    bswizzle<<<16, 256, 0, stream>>>(W3p[l], tmp, 64, 64);
    fill_f16<<<8, 256, 0, stream>>>(tmp, 2048);
    pack_f16<<<7, 256, 0, stream>>>(tmp, 32, 0, 0, po0w, 50, 32);
    bswizzle<<<8, 256, 0, stream>>>(Wo1p[l], tmp, 64, 32);
    fill_f16<<<4, 256, 0, stream>>>(tmp, 1024);
    pack_f16<<<3, 256, 0, stream>>>(tmp, 32, 0, 0, po1w, 32, 18);
    bswizzle<<<4, 256, 0, stream>>>(Wo2p[l], tmp, 32, 32);
  }
  // ---- mlp3 weight packing (phi 0..2 at leaf base 95, psi 0..2 at 113)
  for (int s = 0; s < 6; ++s) {
    const int b = (s < 3) ? (P0 + 92 + s * 6) : (P0 + 110 + (s - 3) * 6);
    fill_f16<<<8, 256, 0, stream>>>(tmp, 2048);
    pack_f16<<<5, 256, 0, stream>>>(tmp, 64, 0, 0, F(b + 1), 18, 64);
    bswizzle<<<8, 256, 0, stream>>>(Wm1p[s], tmp, 32, 64);
    pack_f16<<<16, 256, 0, stream>>>(tmp, 64, 0, 0, F(b + 3), 64, 64);
    bswizzle<<<16, 256, 0, stream>>>(Wm2p[s], tmp, 64, 64);
  }

  // ---- accumulator initialization (deterministic each call)
  fill_f32<<<(NATOM*3 + 255)/256, 256, 0, stream>>>(chi0, NATOM*3, 0.f);
  fill_f32<<<14, 256, 0, stream>>>(ft, 64*54, 0.f);
  fill_f32<<<3, 256, 0, stream>>>(chp, 64*9, 0.f);
  fill_f32<<<1, 256, 0, stream>>>(atmean, 192, 0.f);
  fill_f32<<<1, 256, 0, stream>>>(atmax, 192, -3.0e38f);
  fill_f32<<<1, 64, 0, stream>>>(gcnt, 64, 0.f);
  graph_count<<<(NATOM + 255)/256, 256, 0, stream>>>(batch, gcnt);

  const float* xin[3]  = {x0, x1, x2};
  float* xout[3]       = {x1, x2, x3};
  const float* chin[3] = {chi0, chi1, chi2};
  float* chout[3]      = {chi1, chi2, chi3};
  const int psiSel[3]  = {1, 1, 2};   // reference uses psi[1], psi[1], psi[2]

  for (int l = 0; l < 3; ++l) {
    const int B = P0 + 28 * l;
    fill_f32<<<(NATOM*36 + 255)/256, 256, 0, stream>>>(agg, NATOM*36, 0.f);
    fill_f32<<<(NATOM + 255)/256, 256, 0, stream>>>(cnt, NATOM, 0.f);
    edge_kernel<<<(NEDGE + 63)/64, 128, 0, stream>>>(
        xin[l], chin[l], ei, ea, ev,
        W1p[l], b1f[l],
        W2p[l], F(B+26), F(B+18),                 // sg1b, pm1b
        W3p[l], F(B+8), F(B+12),                  // al0b, be0b
        F(B+11), F(B+10), F(B+15), F(B+14),       // al1w, al1b, be1w, be1b
        agg, cnt);
    node_phi_out<<<(NATOM + 63)/64, 128, 0, stream>>>(
        agg, cnt, xin[l], Wo1p[l], F(B+20), Wo2p[l], F(B+22), xout[l], chout[l]);
    pool_ft_ch<<<(NATOM + 255)/256, 256, 0, stream>>>(xout[l], chout[l], batch, ft, chp, l);
    const int bphi = P0 + 92 + l * 6;
    node_mlp3<<<(NATOM + 63)/64, 128, 0, stream>>>(
        xout[l], Wm1p[l], F(bphi+0), Wm2p[l], F(bphi+2), F(bphi+5), F(bphi+4),
        batch, atmax + l * 64, 0);
    const int bpsi = P0 + 110 + psiSel[l] * 6;
    node_mlp3<<<(NATOM + 63)/64, 128, 0, stream>>>(
        xout[l], Wm1p[3 + psiSel[l]], F(bpsi+0), Wm2p[3 + psiSel[l]], F(bpsi+2),
        F(bpsi+5), F(bpsi+4), batch, atmean + l * 64, 1);
  }

  const int M = P0 + 84;   // mlp leaves: [0].b,[0].w,[1].b,[1].w,...
  const int R = P0 + 128;  // reduce leaves
  final_kernel<<<1, 64, 0, stream>>>(
      ft, chp, gcnt, atmax, atmean,
      F(M+1), F(M+0), F(M+3), F(M+2), F(M+5), F(M+4), F(M+7), F(M+6),
      F(R+1), F(R+0), F(R+3), F(R+2),
      (float*)d_out);
}
